// CausalSelfAttention_18863496364686
// MI455X (gfx1250) — matrix operand order repaired and checked by reading, not compiled, not run on hardware
//
#include <hip/hip_runtime.h>
#include <hip/hip_bf16.h>

// ---------- types ----------
typedef _Float16 h16;
typedef __attribute__((ext_vector_type(8)))  _Float16 v8h_t;
typedef __attribute__((ext_vector_type(16))) _Float16 v16h_t;
typedef __attribute__((ext_vector_type(8)))  float    v8f_t;

#define Bc 4
#define Sc 2048
#define Dc 1024
#define Hc 16
#define HDc 64
#define NEG_BIG (-3.0e38f)

__device__ __forceinline__ v8f_t wmma16(v16h_t a, v16h_t b, v8f_t c) {
  // D = A(16x32 f16) x B(32x16 f16) + C(16x16 f32)
  return __builtin_amdgcn_wmma_f32_16x16x32_f16(
      /*neg_a=*/false, a, /*neg_b=*/false, b,
      /*c_mod=*/(short)0, c, /*reuse_a=*/false, /*reuse_b=*/false);
}

__device__ __forceinline__ v16h_t mkfrag(v8h_t lo, v8h_t hi) {
  v16h_t r;
#pragma unroll
  for (int i = 0; i < 8; ++i) { r[i] = lo[i]; r[i + 8] = hi[i]; }
  return r;
}

__device__ __forceinline__ v8f_t vzero8f() {
  v8f_t z;
#pragma unroll
  for (int i = 0; i < 8; ++i) z[i] = 0.0f;
  return z;
}

// ---------- kernel 0a: fp32 -> fp16 cast ----------
__global__ void cast_f32_f16(const float* __restrict__ in, h16* __restrict__ out, int n) {
  int i = blockIdx.x * blockDim.x + threadIdx.x;
  if (i < n) out[i] = (h16)in[i];
}

// ---------- kernel 0b: fp32 [R][C] -> fp16 transposed [C][R] (one-time weight prep) ----------
__global__ __launch_bounds__(256) void castT_f32_f16(const float* __restrict__ in,
                                                     h16* __restrict__ out, int R, int C) {
  __shared__ float tile[32][33];
  const int r0 = blockIdx.y * 32, c0 = blockIdx.x * 32;
  const int tx = threadIdx.x & 31, ty = threadIdx.x >> 5;  // 32x8
#pragma unroll
  for (int i = ty; i < 32; i += 8)
    tile[i][tx] = in[(size_t)(r0 + i) * C + c0 + tx];
  __syncthreads();
#pragma unroll
  for (int i = ty; i < 32; i += 8)
    out[(size_t)(c0 + i) * R + r0 + tx] = (h16)tile[tx][i];
}

// ---------- kernel 1/3: double-buffered WMMA GEMM  C[M,N] = A[M,K] * Wt[N,K]^T + bias ----
// Wt is PRE-TRANSPOSED [N][K]; both tiles stage with contiguous b128 copies.
// mode 0: scatter f16 into Q/K [B,H,S,64] and V [B,H,64,S]    mode 1: fp32 out [M,N]
__global__ __launch_bounds__(128) void gemm_wmma(
    const h16* __restrict__ A, const h16* __restrict__ Wt,
    const float* __restrict__ bias, int Kdim, int mode,
    h16* __restrict__ Qd, h16* __restrict__ Kd, h16* __restrict__ Vd,
    float* __restrict__ outf) {
  __shared__ __align__(16) h16 As[2][64 * 32];  // [buf][row][k]
  __shared__ __align__(16) h16 Bs[2][64 * 32];  // [buf][n][k]

  const int tid = threadIdx.x;
  const int wave = tid >> 5;
  const int lane = tid & 31;
  const int m0 = blockIdx.y * 64;
  const int n0 = blockIdx.x * 64;

  v8f_t acc[4];
#pragma unroll
  for (int t = 0; t < 4; ++t) acc[t] = vzero8f();

  const int offA = (lane & 16) ? 8 : 0;    // A-layout K offset per lane-half
  const int offB = (lane & 16) ? 16 : 0;   // contiguous K offset per lane-half (A and B frags)
  const int mr = wave * 16 + (lane & 15);
  const int lrow = tid >> 1, lch = tid & 1;  // staging: 64 rows x 2 chunks

  const h16* aSrcBase = A + (size_t)(m0 + lrow) * Kdim + lch * 16;
  const h16* bSrcBase = Wt + (size_t)(n0 + lrow) * Kdim + lch * 16;
  const int stIdx = lrow * 32 + lch * 16;

  uint4 ra0, ra1, rb0, rb1;  // staging registers (global loads fly under WMMAs)

  auto loadTile = [&](int k0) {
    const uint4* sa = (const uint4*)(aSrcBase + k0);
    ra0 = sa[0];
    ra1 = sa[1];
    const uint4* sb = (const uint4*)(bSrcBase + k0);
    rb0 = sb[0];
    rb1 = sb[1];
  };
  auto storeTile = [&](int buf) {
    uint4* da = (uint4*)&As[buf][stIdx];
    da[0] = ra0;
    da[1] = ra1;
    uint4* db = (uint4*)&Bs[buf][stIdx];
    db[0] = rb0;
    db[1] = rb1;
  };
  auto compute = [&](int buf) {
    v16h_t afrag = *(const v16h_t*)&As[buf][mr * 32 + offB];
#pragma unroll
    for (int t = 0; t < 4; ++t) {
      const int nn = t * 16 + (lane & 15);
      v16h_t bfrag = *(const v16h_t*)&Bs[buf][nn * 32 + offB];
      acc[t] = wmma16(afrag, bfrag, acc[t]);
    }
  };

  // software pipeline: Kdim is a multiple of 64
  loadTile(0);
  storeTile(0);
  __syncthreads();
  int k0 = 0;
  for (;;) {
    loadTile(k0 + 32);   // next tile -> regs (no wait yet)
    compute(0);          // WMMAs on current tile hide global latency
    storeTile(1);        // wait lands here, after compute
    __syncthreads();
    const bool more = (k0 + 64 < Kdim);
    if (more) loadTile(k0 + 64);
    compute(1);
    if (!more) break;
    storeTile(0);
    __syncthreads();
    k0 += 64;
  }

  // epilogue (C layout: vgpr r = row r, upper lanes rows 8..15)
  const int lm8 = (lane & 16) ? 8 : 0;
#pragma unroll
  for (int t = 0; t < 4; ++t) {
#pragma unroll
    for (int r = 0; r < 8; ++r) {
      const int m = m0 + wave * 16 + r + lm8;
      const int c = n0 + t * 16 + (lane & 15);
      float val = acc[t][r] + bias[c];
      if (mode == 0) {
        const int which = c >> 10;  // 0=q 1=k 2=v
        const int d = c & 1023;
        const int hh = d >> 6, hd = d & 63;
        const int b = m >> 11, s = m & 2047;
        const size_t bhh = (size_t)(b * Hc + hh);
        if (which == 0)      Qd[(bhh * Sc + s) * HDc + hd] = (h16)val;
        else if (which == 1) Kd[(bhh * Sc + s) * HDc + hd] = (h16)val;
        else                 Vd[(bhh * HDc + hd) * Sc + s] = (h16)val;  // V transposed
      } else {
        outf[(size_t)m * Dc + c] = val;
      }
    }
  }
}

// ---------- kernel 2: causal flash attention (barrier-free kv loop) ----------
// Q,K: [B,H,S,64] f16.  Vt: [B,H,64,S] f16.  Y: [B,S,H,64] f16.
__global__ __launch_bounds__(128) void flash_attn(
    const h16* __restrict__ Q, const h16* __restrict__ K,
    const h16* __restrict__ Vt, h16* __restrict__ Y) {
  __shared__ __align__(16) h16 Pl[4 * 16 * 32];  // per-wave P staging (C->A reshape)

  const int tid = threadIdx.x;
  const int wave = tid >> 5;
  const int lane = tid & 31;
  const int bh = blockIdx.y;  // b*H + h
  const int qb = blockIdx.x;  // 64-row q block

  const h16* Qb = Q + (size_t)bh * Sc * HDc;
  const h16* Kb = K + (size_t)bh * Sc * HDc;
  const h16* Vb = Vt + (size_t)bh * HDc * Sc;  // [64][2048]

  const int q0 = qb * 64 + wave * 16;
  const int offA = (lane & 16) ? 8 : 0;
  const int offB = (lane & 16) ? 16 : 0;
  const int lm8 = (lane & 16) ? 8 : 0;
  const float scale = 0.125f;  // 1/sqrt(64)

  // Q fragments for both 32-wide K chunks (HD=64)
  const h16* qp = Qb + (size_t)(q0 + (lane & 15)) * HDc;
  v16h_t qf[2];
#pragma unroll
  for (int c = 0; c < 2; ++c) {
    qf[c] = *(const v16h_t*)(qp + c * 32 + offB);
  }

  v8f_t O[4];
#pragma unroll
  for (int t = 0; t < 4; ++t) O[t] = vzero8f();
  float mrow[8], lrow[8];
#pragma unroll
  for (int r = 0; r < 8; ++r) { mrow[r] = NEG_BIG; lrow[r] = 0.0f; }

  // each wave runs only to its own causal bound (no block barriers needed)
  const int nkv = (q0 + 15) / 32 + 1;
  for (int j = 0; j < nkv; ++j) {
    const int kb = j * 32;

    if (j + 1 < nkv) {  // prefetch next kv tile rows (global_prefetch_b8)
      __builtin_prefetch(Kb + (size_t)(kb + 32 + (lane & 15)) * HDc, 0, 1);
      __builtin_prefetch(Vb + (size_t)(lane & 31) * Sc + kb + 32, 0, 1);
    }

    // S = Q K^T : two 16x16 score tiles (keys kb..kb+15, kb+16..kb+31)
    v8f_t C0 = vzero8f(), C1 = vzero8f();
    const int keyn = kb + (lane & 15);
#pragma unroll
    for (int c = 0; c < 2; ++c) {
      // B fragment = contiguous 32B run of a key row (K^T column), same k order as qf (k = 16h + s)
      v16h_t kf0 = *(const v16h_t*)(Kb + (size_t)keyn * HDc + c * 32 + offB);
      v16h_t kf1 = *(const v16h_t*)(Kb + (size_t)(keyn + 16) * HDc + c * 32 + offB);
      C0 = wmma16(qf[c], kf0, C0);
      C1 = wmma16(qf[c], kf1, C1);
    }

    // scale + causal mask + online softmax
    float mloc[8], sloc[8], alpha[8];
#pragma unroll
    for (int r = 0; r < 8; ++r) {
      const int qr = q0 + r + lm8;
      float s0 = (keyn <= qr) ? C0[r] * scale : NEG_BIG;
      float s1 = (keyn + 16 <= qr) ? C1[r] * scale : NEG_BIG;
      C0[r] = s0; C1[r] = s1;
      mloc[r] = fmaxf(s0, s1);
    }
#pragma unroll
    for (int msk = 1; msk < 16; msk <<= 1)
#pragma unroll
      for (int r = 0; r < 8; ++r)
        mloc[r] = fmaxf(mloc[r], __shfl_xor(mloc[r], msk, 32));

#pragma unroll
    for (int r = 0; r < 8; ++r) {
      const float mn = fmaxf(mrow[r], mloc[r]);
      alpha[r] = __expf(mrow[r] - mn);
      mrow[r] = mn;
      const float p0 = __expf(C0[r] - mn);
      const float p1 = __expf(C1[r] - mn);
      sloc[r] = p0 + p1;
      const int M = r + lm8;
      Pl[wave * 512 + M * 32 + (lane & 15)] = (h16)p0;       // C-layout -> A-layout
      Pl[wave * 512 + M * 32 + (lane & 15) + 16] = (h16)p1;  // via per-wave LDS slab
    }
#pragma unroll
    for (int msk = 1; msk < 16; msk <<= 1)
#pragma unroll
      for (int r = 0; r < 8; ++r)
        sloc[r] += __shfl_xor(sloc[r], msk, 32);
#pragma unroll
    for (int r = 0; r < 8; ++r) lrow[r] = lrow[r] * alpha[r] + sloc[r];
#pragma unroll
    for (int t = 0; t < 4; ++t)
#pragma unroll
      for (int r = 0; r < 8; ++r) O[t][r] *= alpha[r];

    // O += P(16x32) * V(32x64); V fragments read straight from transposed global V
    v16h_t pf = *(const v16h_t*)&Pl[wave * 512 + (lane & 15) * 32 + offB];
#pragma unroll
    for (int t = 0; t < 4; ++t) {
      v16h_t vf = *(const v16h_t*)(Vb + (size_t)((lane & 15) + 16 * t) * Sc + kb + offB);
      O[t] = wmma16(pf, vf, O[t]);
    }
  }

  // normalize + write Y [B,S,H,64] f16
  const int b = bh >> 4, hh = bh & 15;
#pragma unroll
  for (int t = 0; t < 4; ++t) {
#pragma unroll
    for (int r = 0; r < 8; ++r) {
      const int s = q0 + r + lm8;
      const int n = t * 16 + (lane & 15);
      const float val = O[t][r] / lrow[r];
      Y[(((size_t)b * Sc + s) * Hc + hh) * HDc + n] = (h16)val;
    }
  }
}

// ---------- host ----------
extern "C" void kernel_launch(void* const* d_in, const int* in_sizes, int n_in,
                              void* d_out, int out_size, void* d_ws, size_t ws_size,
                              hipStream_t stream) {
  const float* x      = (const float*)d_in[0];  // [B,S,D]
  const float* W_attn = (const float*)d_in[1];  // [D,3D]
  const float* b_attn = (const float*)d_in[2];  // [3D]
  const float* W_proj = (const float*)d_in[3];  // [D,D]
  const float* b_proj = (const float*)d_in[4];  // [D]
  float* out = (float*)d_out;                   // [B,S,D] fp32

  const size_t nX  = (size_t)Bc * Sc * Dc;       // 8388608
  const size_t nWa = (size_t)Dc * 3 * Dc;        // 3145728
  const size_t nWp = (size_t)Dc * Dc;            // 1048576
  const size_t nQ  = (size_t)Bc * Hc * Sc * HDc; // 8388608

  char* ws = (char*)d_ws;
  h16* x16   = (h16*)(ws);
  h16* Wa16t = (h16*)(ws + nX * 2);                // [3D][D] transposed
  h16* Wp16t = (h16*)(ws + (nX + nWa) * 2);        // [D][D] transposed
  h16* Q16   = (h16*)(ws + (nX + nWa + nWp) * 2);  // [B,H,S,64]
  h16* K16   = Q16 + nQ;                           // [B,H,S,64]
  h16* V16t  = K16 + nQ;                           // [B,H,64,S]
  h16* Y16   = V16t + nQ;                          // [B,S,H,64]; total ~92 MB

  cast_f32_f16<<<(int)((nX + 255) / 256), 256, 0, stream>>>(x, x16, (int)nX);
  {
    dim3 g(3 * Dc / 32, Dc / 32);  // 96 x 32
    castT_f32_f16<<<g, 256, 0, stream>>>(W_attn, Wa16t, Dc, 3 * Dc);
  }
  {
    dim3 g(Dc / 32, Dc / 32);  // 32 x 32
    castT_f32_f16<<<g, 256, 0, stream>>>(W_proj, Wp16t, Dc, Dc);
  }

  // QKV projection: [8192,1024] x [1024,3072] -> Q/K head-major, V transposed
  {
    dim3 grid(3 * Dc / 64, (Bc * Sc) / 64);  // 48 x 128
    gemm_wmma<<<grid, 128, 0, stream>>>(x16, Wa16t, b_attn, Dc, 0,
                                        Q16, K16, V16t, nullptr);
  }

  // causal flash attention
  {
    dim3 grid(Sc / 64, Bc * Hc);  // 32 x 64
    flash_attn<<<grid, 128, 0, stream>>>(Q16, K16, V16t, Y16);
  }

  // output projection: [8192,1024] x [1024,1024] -> fp32 out
  {
    dim3 grid(Dc / 64, (Bc * Sc) / 64);  // 16 x 128
    gemm_wmma<<<grid, 128, 0, stream>>>(Y16, Wp16t, b_proj, Dc, 1,
                                        nullptr, nullptr, nullptr, out);
  }
}
